// shared_lstm_layer_21414706938546
// MI455X (gfx1250) — matrix operand order repaired and checked by reading, not compiled, not run on hardware
//
#include <hip/hip_runtime.h>
#include <hip/hip_bf16.h>
#include <math.h>

// ---------------------------------------------------------------------------
// Bi-LSTM x2 + mean-pool + dense, tuned for MI455X (gfx1250, wave32, WMMA).
//
//  Phase 1: pack all weights into bf16 WMMA-fragment layouts (ISA 7.12.2).
//  Phase 2: big time-parallel GEMMs xz = x @ Wx + b  (v_wmma_f32_16x16x32_bf16,
//           software-pipelined fragment loads, NT stores of xz).
//  Phase 3: persistent recurrent scan kernel: 32 WGs (16 per direction),
//           Wh slice resident in LDS (128KB of the 320KB WGP LDS),
//           per-step h staged with GLOBAL_LOAD_ASYNC_TO_LDS_B128 (ASYNCcnt),
//           grid barrier per timestep via monotonic global atomic.
//           xz added in the gate phase (coalesced) to keep the WMMA loop's
//           register pressure low (no 64-bit addr chains across fragments).
//  Phase 4: layer-2 GEMMs (bf16 A from layer-1 output), second scan.
//  Phase 5: mean over T, dense + ReLU.
// ---------------------------------------------------------------------------

typedef __bf16 bf16;
typedef bf16  bf16x16 __attribute__((ext_vector_type(16)));
typedef float v8f     __attribute__((ext_vector_type(8)));

union Frag16 { uint4 u[2]; bf16x16 v; };

#define BATCH 64
#define SEQT  512
#define HID   512
#define GATES 2048   // 4*HID
#define NWG_SCAN 32  // 16 per direction

__device__ __forceinline__ float sigf(float x) { return 1.0f / (1.0f + __expf(-x)); }

// ---- CDNA5 async global->LDS (bypasses VGPRs, tracked by ASYNCcnt) --------
__device__ __forceinline__ void async_ld_b128(const void* gptr, void* lptr) {
  unsigned lds = (unsigned)(unsigned long long)lptr;   // low 32b of flat = LDS offset
  unsigned long long g = (unsigned long long)gptr;
  asm volatile("global_load_async_to_lds_b128 %0, %1, off"
               :: "v"(lds), "v"(g) : "memory");
}
__device__ __forceinline__ void wait_async0() {
  asm volatile("s_wait_asynccnt 0x0" ::: "memory");
}

// ---------------------------------------------------------------------------
// Weight packing into WMMA fragment order.
// A-frag (16x32 bf16): lane half hA=lane>>4 : k = (j>>3)*16 + hA*8 + (j&7)
// B-frag (32x16 bf16): lane half hB=lane>>4 : k = (j>>3)*16 + hB*8 + (j&7) ; n = lane&15
// ---------------------------------------------------------------------------

// Pack Wx [K x 2048] -> frags indexed [nt(0..127)][kt(0..KT-1)][lane][j]
__global__ void pack_wx_kernel(const float* __restrict__ W, bf16* __restrict__ out,
                               int KT, int n_total) {
  int e = blockIdx.x * 256 + threadIdx.x;
  if (e >= n_total) return;
  int j    = e & 15;
  int lane = (e >> 4) & 31;
  int frag = e >> 9;              // nt*KT + kt
  int kt   = frag % KT;
  int nt   = frag / KT;
  int k = kt * 32 + (j >> 3) * 16 + (lane >> 4) * 8 + (j & 7);
  int n = nt * 16 + (lane & 15);
  out[e] = (bf16)W[(size_t)k * GATES + n];
}

// Pack Wh [512 x 2048] -> frags grouped per scan-workgroup slice:
//   [g(0..15)][lt(0..7)][kt(0..15)][lane][j]
//   lt -> gate = lt>>1, nt2 = lt&1 ; n = gate*512 + g*32 + nt2*16 + (lane&15)
__global__ void pack_wh_kernel(const float* __restrict__ W, bf16* __restrict__ out) {
  int e = blockIdx.x * 256 + threadIdx.x;   // 1,048,576 total
  int j    = e & 15;
  int lane = (e >> 4) & 31;
  int kt   = (e >> 9) & 15;
  int lt   = (e >> 13) & 7;
  int g    = (e >> 16) & 15;
  int gate = lt >> 1, nt2 = lt & 1;
  int n = gate * HID + g * 32 + nt2 * 16 + (lane & 15);
  int k = kt * 32 + (j >> 3) * 16 + (lane >> 4) * 8 + (j & 7);
  out[e] = (bf16)W[(size_t)k * GATES + n];
}

__global__ void init_ctr_kernel(unsigned* ctr) {
  if (threadIdx.x < 64) ctr[threadIdx.x] = 0u;
}

// ---------------------------------------------------------------------------
// Time-parallel GEMM: out[M=32768, N=2048] = A[M,K] @ B[K,N] + bias
// Block tile 64x128, 8 waves, each wave: fixed mt (16 rows), 4 n-tiles.
// Two-stage software pipeline over kt so frag loads overlap WMMA issue.
// ---------------------------------------------------------------------------
template <bool A_IS_F32, int K>
__global__ void gemm_xz_kernel(const void* __restrict__ Aptr,
                               const bf16* __restrict__ Bp,
                               const float* __restrict__ bias,
                               float* __restrict__ out) {
  extern __shared__ char smem[];
  bf16* As = (bf16*)smem;
  constexpr int SA = K + 8;             // padded row stride (conflict-free b128)
  constexpr int KT = K / 32;
  const int tid = threadIdx.x;
  const int mblock = blockIdx.x;        // 512
  const int nblock = blockIdx.y;        // 16

  // ---- stage A tile [64 x K] into LDS as bf16 ----
  if (A_IS_F32) {
    const float* A = (const float*)Aptr;
    constexpr int nvec = (64 * K) / 4;
    for (int v = tid; v < nvec; v += 256) {
      int flat = v * 4;
      int row = flat / K, k = flat % K;
      float4 a = *(const float4*)(A + (size_t)(mblock * 64 + row) * K + k);
      bf16* d = As + row * SA + k;
      d[0] = (bf16)a.x; d[1] = (bf16)a.y; d[2] = (bf16)a.z; d[3] = (bf16)a.w;
    }
  } else {
    const bf16* A = (const bf16*)Aptr;
    constexpr int nvec = (64 * K) / 8;
    for (int v = tid; v < nvec; v += 256) {
      int flat = v * 8;
      int row = flat / K, k = flat % K;
      async_ld_b128(A + (size_t)(mblock * 64 + row) * K + k, As + row * SA + k);
    }
    wait_async0();
  }
  __syncthreads();

  const int wave = tid >> 5;
  const int lane = tid & 31;
  const int mt = wave & 3;
  const int ntbase = (wave >> 2) * 4;
  const int hA = lane >> 4;

  v8f acc[4];
#pragma unroll
  for (int i = 0; i < 4; ++i) acc[i] = (v8f)(0.0f);

  const bf16* arow = As + (mt * 16 + (lane & 15)) * SA + hA * 8;
  const bf16* bbase = Bp + ((size_t)(nblock * 8 + ntbase) * KT * 32 + lane) * 16;

  auto loadA = [&](Frag16& a, int kt) {
    const bf16* ab = arow + kt * 32;
    a.u[0] = *(const uint4*)ab;
    a.u[1] = *(const uint4*)(ab + 16);
  };
  auto loadB = [&](Frag16* b, int kt) {
#pragma unroll
    for (int i = 0; i < 4; ++i) {
      const bf16* bb = bbase + ((size_t)i * KT + kt) * 512;
      b[i].u[0] = *(const uint4*)bb;
      b[i].u[1] = *(const uint4*)(bb + 8);
    }
  };
  auto mm4 = [&](Frag16& a, Frag16* b) {
#pragma unroll
    for (int i = 0; i < 4; ++i)
      acc[i] = __builtin_amdgcn_wmma_f32_16x16x32_bf16(
          false, a.v, false, b[i].v, (short)0, acc[i], false, false);
  };

  Frag16 a0, a1, b0[4], b1[4];
  loadA(a0, 0); loadB(b0, 0);
#pragma unroll
  for (int kt = 0; kt < KT; kt += 2) {
    if (kt + 1 < KT) { loadA(a1, kt + 1); loadB(b1, kt + 1); }
    mm4(a0, b0);
    if (kt + 1 < KT) {
      if (kt + 2 < KT) { loadA(a0, kt + 2); loadB(b0, kt + 2); }
      mm4(a1, b1);
    }
  }

  // ---- epilogue: add bias, NT-store fp32 (xz streamed, read once) ----
#pragma unroll
  for (int i = 0; i < 4; ++i) {
    int col = (nblock * 8 + ntbase + i) * 16 + (lane & 15);
    float bv = bias[col];
#pragma unroll
    for (int r = 0; r < 8; ++r) {
      int row = mblock * 64 + mt * 16 + r + hA * 8;
      __builtin_nontemporal_store(acc[i][r] + bv, out + (size_t)row * GATES + col);
    }
  }
}

// ---------------------------------------------------------------------------
// Persistent recurrent scan. 32 WGs: dir = blk>>4, g = blk&15 (hidden slice
// g*32..g*32+31). Wh slice (128KB bf16 frags) resident in LDS whole scan.
// Per step: h[64x512] async-copied L2->LDS, 64x v_wmma, gate math, grid barrier.
// ---------------------------------------------------------------------------
#define H_STRIDE 520    // bf16 elems; 1040B rows -> conflict-free b128
#define Z_STRIDE 132

__global__ void lstm_scan_kernel(const float* __restrict__ xz,   // [2][64][512][2048]
                                 const bf16* __restrict__ Whp,   // [2][16][8][16][512]
                                 bf16* __restrict__ hbuf,        // [2][64][512]
                                 bf16* __restrict__ hseq,        // [64][512][1024]
                                 unsigned* __restrict__ ctr) {
  extern __shared__ char smem[];
  bf16*  whp_s = (bf16*)smem;                                   // 131072 B
  bf16*  h_s   = (bf16*)(smem + 131072);                        // 64*520*2 = 66560 B
  float* z_s   = (float*)(smem + 131072 + 66560);               // 64*132*4 = 33792 B
  float* c_s   = (float*)(smem + 131072 + 66560 + 33792);       // 8192 B

  const int tid  = threadIdx.x;
  const int dir  = blockIdx.x >> 4;
  const int g    = blockIdx.x & 15;
  const int wave = tid >> 5;
  const int lane = tid & 31;
  const int mt = wave & 3;
  const int ltbase = (wave >> 2) * 4;
  const int hA = lane >> 4;

  // ---- one-time: Wh slice -> LDS via async DMA, zero cell state ----
  {
    const bf16* src = Whp + ((size_t)(dir * 16 + g)) * 65536;
    for (int it = 0; it < 32; ++it) {
      int idx = (it * 256 + tid) * 8;
      async_ld_b128(src + idx, whp_s + idx);
    }
    for (int q = 0; q < 8; ++q) c_s[q * 256 + tid] = 0.0f;
    wait_async0();
  }

  const int row_cp = tid >> 2;            // h copy: 4 threads per batch row
  const int seg_cp = (tid & 3) * 128;

  const bf16* arow = h_s + (mt * 16 + (lane & 15)) * H_STRIDE + hA * 8;
  const bf16* bbase = whp_s + (ltbase * 16 * 32 + lane) * 16;

  // gate-phase constants: each thread owns 8 (b,hl) elements
  const int ew_b0  = tid >> 5;            // b for q=0 item (idx = q*256+tid)
  const int ew_hl  = tid & 31;

  for (int step = 0; step < SEQT; ++step) {
    const int t = dir ? (SEQT - 1 - step) : step;

    // ---- h (prev step, full hidden) -> LDS ----
    if (step == 0) {
      uint4 z4 = make_uint4(0u, 0u, 0u, 0u);
      for (int i = 0; i < 16; ++i)
        *(uint4*)(h_s + row_cp * H_STRIDE + seg_cp + i * 8) = z4;
    } else {
      const bf16* hsrc = hbuf + (size_t)dir * BATCH * HID + row_cp * HID + seg_cp;
      bf16* hdst = h_s + row_cp * H_STRIDE + seg_cp;
#pragma unroll
      for (int i = 0; i < 16; ++i)
        async_ld_b128(hsrc + i * 8, hdst + i * 8);
      wait_async0();
    }
    __syncthreads();

    // ---- z-slice = h @ Wh_slice  (64 x 128, K=512), pipelined frags ----
    v8f acc[4];
#pragma unroll
    for (int i = 0; i < 4; ++i) acc[i] = (v8f)(0.0f);

    auto loadA = [&](Frag16& a, int kt) {
      const bf16* ab = arow + kt * 32;
      a.u[0] = *(const uint4*)ab;
      a.u[1] = *(const uint4*)(ab + 16);
    };
    auto loadB = [&](Frag16* b, int kt) {
#pragma unroll
      for (int i = 0; i < 4; ++i) {
        const bf16* bb = bbase + (i * 16 + kt) * 512;
        b[i].u[0] = *(const uint4*)bb;
        b[i].u[1] = *(const uint4*)(bb + 8);
      }
    };
    auto mm4 = [&](Frag16& a, Frag16* b) {
#pragma unroll
      for (int i = 0; i < 4; ++i)
        acc[i] = __builtin_amdgcn_wmma_f32_16x16x32_bf16(
            false, a.v, false, b[i].v, (short)0, acc[i], false, false);
    };

    Frag16 a0, a1, b0[4], b1[4];
    loadA(a0, 0); loadB(b0, 0);
#pragma unroll
    for (int kt = 0; kt < 16; kt += 2) {
      if (kt + 1 < 16) { loadA(a1, kt + 1); loadB(b1, kt + 1); }
      mm4(a0, b0);
      if (kt + 1 < 16) {
        if (kt + 2 < 16) { loadA(a0, kt + 2); loadB(b0, kt + 2); }
        mm4(a1, b1);
      }
    }

    // ---- stage accumulators to LDS (pure stores, no address chains) ----
#pragma unroll
    for (int i = 0; i < 4; ++i) {
      int lc = (ltbase + i) * 16 + (lane & 15);
#pragma unroll
      for (int r = 0; r < 8; ++r) {
        int b = mt * 16 + r + hA * 8;
        z_s[b * Z_STRIDE + lc] = acc[i][r];
      }
    }
    __syncthreads();

    // ---- gates (with coalesced xz add here):
    //      c = sig(f+xf)*c + sig(i+xi)*tanh(g+xg); h = sig(o+xo)*tanh(c) ----
#pragma unroll
    for (int q = 0; q < 8; ++q) {
      int idx = q * 256 + tid;          // 2048 = 64 batch x 32 hidden
      int b = ew_b0 + q * 8;            // (q*256+tid)>>5
      int hl = ew_hl;
      const float* xrow =
          xz + (((size_t)dir * BATCH + b) * SEQT + t) * GATES + g * 32 + hl;
      float iv = z_s[b * Z_STRIDE + hl]      + __builtin_nontemporal_load(xrow);
      float fv = z_s[b * Z_STRIDE + 32 + hl] + __builtin_nontemporal_load(xrow + HID);
      float gv = z_s[b * Z_STRIDE + 64 + hl] + __builtin_nontemporal_load(xrow + 2 * HID);
      float ov = z_s[b * Z_STRIDE + 96 + hl] + __builtin_nontemporal_load(xrow + 3 * HID);
      float c = sigf(fv) * c_s[idx] + sigf(iv) * tanhf(gv);
      c_s[idx] = c;
      float h = sigf(ov) * tanhf(c);
      bf16 hb = (bf16)h;
      int hg = g * 32 + hl;
      hbuf[(size_t)dir * BATCH * HID + b * HID + hg] = hb;
      __builtin_nontemporal_store(
          hb, hseq + ((size_t)b * SEQT + t) * (2 * HID) + dir * HID + hg);
    }

    // ---- grid barrier (monotonic counter, agent scope) ----
    __syncthreads();
    if (tid == 0) {
      __hip_atomic_fetch_add(ctr, 1u, __ATOMIC_RELEASE, __HIP_MEMORY_SCOPE_AGENT);
      unsigned target = (unsigned)NWG_SCAN * (unsigned)(step + 1);
      while (__hip_atomic_load(ctr, __ATOMIC_ACQUIRE, __HIP_MEMORY_SCOPE_AGENT) < target)
        __builtin_amdgcn_s_sleep(2);
    }
    __syncthreads();
    __threadfence();   // invalidate WGP-level cached hbuf lines before reload
  }
}

// ---------------------------------------------------------------------------
// Mean over T, then dense + ReLU.
// ---------------------------------------------------------------------------
__global__ void mean_pool_kernel(const bf16* __restrict__ h2, float* __restrict__ avg) {
  int idx = blockIdx.x * 256 + threadIdx.x;   // 64*1024
  int b = idx >> 10, n = idx & 1023;
  float s = 0.0f;
  for (int t = 0; t < SEQT; ++t)
    s += (float)__builtin_nontemporal_load(h2 + ((size_t)b * SEQT + t) * 1024 + n);
  avg[idx] = s * (1.0f / (float)SEQT);
}

__global__ void dense_relu_kernel(const float* __restrict__ avg,
                                  const float* __restrict__ wd,
                                  const float* __restrict__ bd,
                                  float* __restrict__ out) {
  int idx = blockIdx.x * 256 + threadIdx.x;   // 64*512
  int b = idx >> 9, o = idx & 511;
  float s = bd[o];
  for (int k = 0; k < 1024; ++k)
    s = fmaf(avg[b * 1024 + k], wd[(size_t)k * 512 + o], s);
  out[idx] = fmaxf(s, 0.0f);
}

// ---------------------------------------------------------------------------
extern "C" void kernel_launch(void* const* d_in, const int* in_sizes, int n_in,
                              void* d_out, int out_size, void* d_ws, size_t ws_size,
                              hipStream_t stream) {
  const float* x     = (const float*)d_in[0];
  const float* w1f_x = (const float*)d_in[1];
  const float* w1f_h = (const float*)d_in[2];
  const float* b1f   = (const float*)d_in[3];
  const float* w1b_x = (const float*)d_in[4];
  const float* w1b_h = (const float*)d_in[5];
  const float* b1b   = (const float*)d_in[6];
  const float* w2f_x = (const float*)d_in[7];
  const float* w2f_h = (const float*)d_in[8];
  const float* b2f   = (const float*)d_in[9];
  const float* w2b_x = (const float*)d_in[10];
  const float* w2b_h = (const float*)d_in[11];
  const float* b2b   = (const float*)d_in[12];
  const float* wd    = (const float*)d_in[13];
  const float* bd    = (const float*)d_in[14];
  float* out = (float*)d_out;

  // ---- workspace carve-up ----
  size_t off = 0;
  char* ws = (char*)d_ws;
  auto carve = [&](size_t bytes) -> void* {
    void* p = ws + off;
    off = (off + bytes + 255) & ~(size_t)255;
    return p;
  };
  unsigned* ctr  = (unsigned*)carve(256);
  bf16* Wxp1f = (bf16*)carve((size_t)256 * GATES * 2);
  bf16* Wxp1b = (bf16*)carve((size_t)256 * GATES * 2);
  bf16* Wxp2f = (bf16*)carve((size_t)1024 * GATES * 2);
  bf16* Wxp2b = (bf16*)carve((size_t)1024 * GATES * 2);
  bf16* Whp1  = (bf16*)carve((size_t)2 * HID * GATES * 2);
  bf16* Whp2  = (bf16*)carve((size_t)2 * HID * GATES * 2);
  bf16* hbuf  = (bf16*)carve((size_t)2 * BATCH * HID * 2);
  float* avg  = (float*)carve((size_t)BATCH * 1024 * 4);
  bf16* h1cat = (bf16*)carve((size_t)BATCH * SEQT * 1024 * 2);
  bf16* h2cat = (bf16*)carve((size_t)BATCH * SEQT * 1024 * 2);
  float* xz   = (float*)carve((size_t)2 * BATCH * SEQT * GATES * 4);  // reused L1/L2

  // dynamic LDS sizes
  const int smem_gemm1 = 64 * (256 + 8) * 2;      //  33,792 B
  const int smem_gemm2 = 64 * (1024 + 8) * 2;     // 132,096 B
  const int smem_scan  = 131072 + 66560 + 33792 + 8192;  // 239,616 B
  hipFuncSetAttribute((const void*)gemm_xz_kernel<true, 256>,
                      hipFuncAttributeMaxDynamicSharedMemorySize, smem_gemm1);
  hipFuncSetAttribute((const void*)gemm_xz_kernel<false, 1024>,
                      hipFuncAttributeMaxDynamicSharedMemorySize, smem_gemm2);
  hipFuncSetAttribute((const void*)lstm_scan_kernel,
                      hipFuncAttributeMaxDynamicSharedMemorySize, smem_scan);

  // ---- phase 0: counters + weight packing ----
  init_ctr_kernel<<<1, 64, 0, stream>>>(ctr);
  {
    int n1 = 256 * GATES;                       // 524,288
    pack_wx_kernel<<<n1 / 256, 256, 0, stream>>>(w1f_x, Wxp1f, 8, n1);
    pack_wx_kernel<<<n1 / 256, 256, 0, stream>>>(w1b_x, Wxp1b, 8, n1);
    int n2 = 1024 * GATES;                      // 2,097,152
    pack_wx_kernel<<<n2 / 256, 256, 0, stream>>>(w2f_x, Wxp2f, 32, n2);
    pack_wx_kernel<<<n2 / 256, 256, 0, stream>>>(w2b_x, Wxp2b, 32, n2);
    pack_wh_kernel<<<4096, 256, 0, stream>>>(w1f_h, Whp1);
    pack_wh_kernel<<<4096, 256, 0, stream>>>(w1b_h, Whp1 + (size_t)HID * GATES);
    pack_wh_kernel<<<4096, 256, 0, stream>>>(w2f_h, Whp2);
    pack_wh_kernel<<<4096, 256, 0, stream>>>(w2b_h, Whp2 + (size_t)HID * GATES);
  }

  const size_t xz_dir_stride = (size_t)BATCH * SEQT * GATES;
  dim3 ggrid(512, 16);

  // ---- layer 1: input projections + scan ----
  gemm_xz_kernel<true, 256><<<ggrid, 256, smem_gemm1, stream>>>(x, Wxp1f, b1f, xz);
  gemm_xz_kernel<true, 256><<<ggrid, 256, smem_gemm1, stream>>>(x, Wxp1b, b1b,
                                                                xz + xz_dir_stride);
  lstm_scan_kernel<<<NWG_SCAN, 256, smem_scan, stream>>>(xz, Whp1, hbuf, h1cat, ctr + 0);

  // ---- layer 2: input projections (A = bf16 h1cat) + scan ----
  gemm_xz_kernel<false, 1024><<<ggrid, 256, smem_gemm2, stream>>>(h1cat, Wxp2f, b2f, xz);
  gemm_xz_kernel<false, 1024><<<ggrid, 256, smem_gemm2, stream>>>(h1cat, Wxp2b, b2b,
                                                                  xz + xz_dir_stride);
  lstm_scan_kernel<<<NWG_SCAN, 256, smem_scan, stream>>>(xz, Whp2, hbuf, h2cat, ctr + 1);

  // ---- head: mean over T, dense + ReLU ----
  mean_pool_kernel<<<(BATCH * 1024) / 256, 256, 0, stream>>>(h2cat, avg);
  dense_relu_kernel<<<(BATCH * 512) / 256, 256, 0, stream>>>(avg, wd, bd, out);
}
